// QueryProposal_74947179315960
// MI455X (gfx1250) — matrix operand order repaired and checked by reading, not compiled, not run on hardware
//
#include <hip/hip_runtime.h>
#include <hip/hip_bf16.h>
#include <stdint.h>

// ---------------------------------------------------------------------------
// Types for CDNA5 WMMA (wave32): 16x16x32 bf16 -> f32
// ---------------------------------------------------------------------------
typedef __bf16 v16bf __attribute__((ext_vector_type(16)));
typedef __bf16 v8bf  __attribute__((ext_vector_type(8)));
typedef float  v8f   __attribute__((ext_vector_type(8)));
typedef unsigned int u32x4 __attribute__((ext_vector_type(4)));
typedef unsigned int u32x2 __attribute__((ext_vector_type(2)));

union B128u { u32x4 u; v8bf h; };

__device__ inline v8f zero_v8f() {
  v8f z;
#pragma unroll
  for (int i = 0; i < 8; ++i) z[i] = 0.f;
  return z;
}

__device__ inline unsigned short f2bf(float f) {
  unsigned int u = __float_as_uint(f);
  u += 0x7FFFu + ((u >> 16) & 1u);   // round-to-nearest-even
  return (unsigned short)(u >> 16);
}

enum : int { Bn = 8, Cn = 256, Hn = 96, Wn = 96, HW = 9216, NQn = 100, OBJ = 11 };

// ---------------------------------------------------------------------------
// Packing: NCHW f32 -> NHWC bf16  (K = (tap, cin) becomes contiguous in cin)
// ---------------------------------------------------------------------------
__global__ void qp_nchw_to_nhwc_bf16(const float* __restrict__ in,
                                     unsigned short* __restrict__ out,
                                     int C, int total) {
  int e = blockIdx.x * blockDim.x + threadIdx.x;
  if (e >= total) return;
  int c = e % C;
  int w = (e / C) % Wn;
  int h = (e / (C * Wn)) % Hn;
  int b = e / (C * Wn * Hn);
  out[e] = f2bf(in[(((size_t)b * C + c) * Hn + h) * Wn + w]);
}

// Weights [Cout][Cin][3][3] f32 -> exact per-lane B-fragment layout bf16:
//   [kc][nt][lane][j] with K = kc*32 + ((lane&16)?16:0) + j, cout = nt*16 + (lane&15)
//   K global order = tap*CIN + cin
__global__ void qp_pack_wfrag(const float* __restrict__ w,
                              unsigned short* __restrict__ out,
                              int CIN, int COUT, int total) {
  int e = blockIdx.x * blockDim.x + threadIdx.x;
  if (e >= total) return;
  int j    = e & 15;
  int lane = (e >> 4) & 31;
  int rest = e >> 9;
  int NT   = COUT >> 4;
  int nt   = rest % NT;
  int kc   = rest / NT;
  int K    = kc * 32 + ((lane & 16) ? 16 : 0) + j;
  int cout = nt * 16 + (lane & 15);
  int tap  = K / CIN;
  int cin  = K % CIN;
  out[e] = f2bf(w[((size_t)cout * CIN + cin) * 9 + tap]);
}

// ---------------------------------------------------------------------------
// Implicit-GEMM 3x3 conv with WMMA bf16 (zero-pad=1).
// A-tile staged into LDS with CDNA5 async global->LDS DMA (ASYNCcnt),
// double-buffered: DMA for chunk k+1 overlaps WMMAs of chunk k; one
// workgroup barrier per K-chunk.  B fragments stream from global with
// offset-based addressing (keeps GLOBAL encoding -> LOADcnt only, no
// DScnt coupling) plus global_prefetch of the next chunk.
// MODE 0: fused bias+ReLU+1x1 conv (OC2 outs) -> f32 NCHW logits
// MODE 1: bias+ReLU -> bf16 NHWC (feeds next conv)
// MODE 2: bias+ReLU -> f32 NCHW
// Grid: (H*(W/32), B), block 256 (8 waves). Tile: 32 positions x COUT.
// ---------------------------------------------------------------------------
template <int CIN, int COUT, int MODE, int OC2>
__global__ __launch_bounds__(256) void qp_conv3x3_wmma(
    const unsigned short* __restrict__ xb,     // [B,H,W,CIN] bf16
    const unsigned short* __restrict__ wfrag,  // swizzled weights bf16
    const float* __restrict__ bias1,           // [COUT]
    const float* __restrict__ w2,              // [OC2][COUT] (MODE0)
    const float* __restrict__ b2,              // [OC2]      (MODE0)
    void* __restrict__ outp) {
  constexpr int NT  = COUT / 16;
  constexpr int NB  = (NT > 8) ? NT / 8 : 1;   // n-tiles per wave
  constexpr int CCH = CIN / 32;                // cin chunks per tap
  constexpr int KC  = 9 * CCH;                 // total K chunks (K = 9*CIN)
  constexpr int WSTRIDE = NT * 32 * 16;        // wfrag elements per K-chunk

  const int tid  = threadIdx.x;
  const int wave = tid >> 5;
  const int lane = tid & 31;
  const int bx   = blockIdx.x;
  const int h    = bx / 3;
  const int w0   = (bx % 3) * 32;
  const int b    = blockIdx.y;

  __shared__ __align__(16) unsigned short Abuf[2][32][40];  // double-buffered A tile

  v8f acc[2][NB];
#pragma unroll
  for (int mb = 0; mb < 2; ++mb)
#pragma unroll
    for (int i = 0; i < NB; ++i) acc[mb][i] = zero_v8f();

  const bool active = (NT > 8) || (wave < NT);

  // cooperative A-stage mapping: 256 thr -> 32 rows x 8 groups of 4 cin (8B each)
  const int lm  = tid >> 3;
  const int lco = (tid & 7) * 4;
  // A layout (matches B frag): lane<16 -> K{0..15}; lane>=16 -> K{16..31}
  const int klo_a = (lane & 16) ? 16 : 0;

  // running weight-fragment OFFSETS off the kernarg pointer: keeps the GEP
  // rooted at the __restrict__ global arg so loads stay GLOBAL (not FLAT)
  size_t woff[NB];
#pragma unroll
  for (int i = 0; i < NB; ++i) {
    const int nt = (NT > 8) ? (wave * NB + i) : wave;
    woff[i] = ((size_t)nt * 32 + lane) * 16;
  }

  // stage K-chunk kc into LDS buffer s via async global->LDS DMA;
  // out-of-bounds halo lanes zero-fill with a plain ds_store (no race:
  // those lanes issue no async write to their slot).
  auto stage = [&](int kc, int s) {
    const int tap = kc / CCH;
    const int c0  = (kc % CCH) * 32;
    const int dy  = tap / 3 - 1;
    const int dx  = tap % 3 - 1;
    const int hh  = h + dy;
    const int ww  = w0 + lm + dx;
    unsigned short* lp = &Abuf[s][lm][lco];
    if ((unsigned)hh < 96u && (unsigned)ww < 96u) {
      const size_t off = (((size_t)b * 96 + hh) * 96 + ww) * CIN + c0 + lco;
      const unsigned long long ga = (unsigned long long)(xb + off);
      const unsigned ldsoff = (unsigned)(unsigned long long)(uintptr_t)lp;
      asm volatile("global_load_async_to_lds_b64 %0, %1, off"
                   :: "v"(ldsoff), "v"(ga)
                   : "memory");
    } else {
      u32x2 z; z[0] = 0u; z[1] = 0u;
      *(u32x2*)lp = z;
    }
  };

  // prologue: stage chunk 0, drain ASYNCcnt, sync workgroup
  stage(0, 0);
  asm volatile("s_wait_asynccnt 0" ::: "memory");
  __syncthreads();

  for (int kc = 0; kc < KC; ++kc) {
    const int s = kc & 1;
    if (kc + 1 < KC) stage(kc + 1, s ^ 1);   // overlap DMA with compute

    // A fragments from LDS (ds_load_b128 x2 per m-block)
    v16bf afrag[2];
#pragma unroll
    for (int mb = 0; mb < 2; ++mb) {
      const int m = mb * 16 + (lane & 15);
      B128u lo, hi;
      lo.u = *(const u32x4*)(&Abuf[s][m][klo_a]);
      hi.u = *(const u32x4*)(&Abuf[s][m][klo_a + 8]);
#pragma unroll
      for (int j = 0; j < 8; ++j) { afrag[mb][j] = lo.h[j]; afrag[mb][j + 8] = hi.h[j]; }
    }

    if (active) {
#pragma unroll
      for (int i = 0; i < NB; ++i) {
        if (kc + 1 < KC)
          __builtin_prefetch(wfrag + woff[i] + WSTRIDE, 0, 3);
        B128u blo, bhi;
        blo.u = *(const u32x4*)(wfrag + woff[i]);
        bhi.u = *(const u32x4*)(wfrag + woff[i] + 8);
        woff[i] += WSTRIDE;
        v16bf bfrag;
#pragma unroll
        for (int j = 0; j < 8; ++j) { bfrag[j] = blo.h[j]; bfrag[j + 8] = bhi.h[j]; }
#pragma unroll
        for (int mb = 0; mb < 2; ++mb)
          acc[mb][i] = __builtin_amdgcn_wmma_f32_16x16x32_bf16(
              false, afrag[mb], false, bfrag, (short)0, acc[mb][i], false, false);
      }
    }

    // wave's async DMAs for chunk kc+1 must land before anyone reads them
    asm volatile("s_wait_asynccnt 0" ::: "memory");
    __syncthreads();
  }

  // ---------------- epilogue ----------------
  if constexpr (MODE == 0) {
    __shared__ float red[32][OC2];
    for (int t = tid; t < 32 * OC2; t += 256) (&red[0][0])[t] = 0.f;
    __syncthreads();
    if (active) {
#pragma unroll
      for (int i = 0; i < NB; ++i) {
        const int nt   = (NT > 8) ? (wave * NB + i) : wave;
        const int cout = nt * 16 + (lane & 15);
        const float bb = bias1[cout];
#pragma unroll
        for (int mb = 0; mb < 2; ++mb) {
#pragma unroll
          for (int r = 0; r < 8; ++r) {
            const int m = mb * 16 + r + ((lane & 16) ? 8 : 0);
            float v = acc[mb][i][r] + bb;
            v = v > 0.f ? v : 0.f;
#pragma unroll
            for (int oc = 0; oc < OC2; ++oc) {
              float s = v * w2[oc * COUT + cout];
              s += __shfl_xor(s, 1);
              s += __shfl_xor(s, 2);
              s += __shfl_xor(s, 4);
              s += __shfl_xor(s, 8);
              if ((lane & 15) == 0) atomicAdd(&red[m][oc], s);
            }
          }
        }
      }
    }
    __syncthreads();
    float* out = (float*)outp;
    for (int t = tid; t < 32 * OC2; t += 256) {
      const int m = t / OC2, oc = t % OC2;
      out[(((size_t)b * OC2 + oc) * 96 + h) * 96 + (w0 + m)] = red[m][oc] + b2[oc];
    }
  } else if constexpr (MODE == 1) {
    if (active) {
      unsigned short* out = (unsigned short*)outp;
#pragma unroll
      for (int i = 0; i < NB; ++i) {
        const int nt   = (NT > 8) ? (wave * NB + i) : wave;
        const int cout = nt * 16 + (lane & 15);
        const float bb = bias1[cout];
#pragma unroll
        for (int mb = 0; mb < 2; ++mb)
#pragma unroll
          for (int r = 0; r < 8; ++r) {
            const int m = mb * 16 + r + ((lane & 16) ? 8 : 0);
            float v = acc[mb][i][r] + bb;
            v = v > 0.f ? v : 0.f;
            out[(((size_t)b * 96 + h) * 96 + (w0 + m)) * COUT + cout] = f2bf(v);
          }
      }
    }
  } else {
    if (active) {
      float* out = (float*)outp;
#pragma unroll
      for (int i = 0; i < NB; ++i) {
        const int nt   = (NT > 8) ? (wave * NB + i) : wave;
        const int cout = nt * 16 + (lane & 15);
        const float bb = bias1[cout];
#pragma unroll
        for (int mb = 0; mb < 2; ++mb)
#pragma unroll
          for (int r = 0; r < 8; ++r) {
            const int m = mb * 16 + r + ((lane & 16) ? 8 : 0);
            float v = acc[mb][i][r] + bb;
            v = v > 0.f ? v : 0.f;
            out[(((size_t)b * COUT + cout) * 96 + h) * 96 + (w0 + m)] = v;
          }
      }
    }
  }
}

// ---------------------------------------------------------------------------
// softmax over channels + argmax one-hot mask over foreground channels
// ---------------------------------------------------------------------------
template <int OC>
__global__ void qp_softmax_mask(const float* __restrict__ logits,
                                float* __restrict__ pout) {
  int e = blockIdx.x * blockDim.x + threadIdx.x;
  if (e >= Bn * HW) return;
  int b = e / HW, pix = e % HW;
  const float* L = logits + (size_t)b * OC * HW + pix;
  float v[OC], mx = -1e30f;
#pragma unroll
  for (int ch = 0; ch < OC; ++ch) { v[ch] = L[(size_t)ch * HW]; mx = fmaxf(mx, v[ch]); }
  float s = 0.f;
#pragma unroll
  for (int ch = 0; ch < OC; ++ch) { v[ch] = __expf(v[ch] - mx); s += v[ch]; }
  const float inv = 1.f / s;
  int am = 0; float bv = v[0];
#pragma unroll
  for (int ch = 1; ch < OC - 1; ++ch)
    if (v[ch] > bv) { bv = v[ch]; am = ch; }
#pragma unroll
  for (int ch = 0; ch < OC; ++ch)
    pout[((size_t)b * OC + ch) * HW + pix] = (ch == am) ? v[ch] * inv : 0.f;
}

// 8-neighbor local max (zero pad), score = p + 1[localmax]; reduce max over NCH
template <int OC, int NCH>
__global__ void qp_nms_score(const float* __restrict__ p, float* __restrict__ score) {
  int e = blockIdx.x * blockDim.x + threadIdx.x;
  if (e >= Bn * HW) return;
  int b = e / HW, pix = e % HW;
  int hh = pix / Wn, ww = pix % Wn;
  float best = -1e30f;
  for (int ch = 0; ch < NCH; ++ch) {
    const float* pc = p + ((size_t)b * OC + ch) * HW;
    float pv = pc[pix];
    bool m = (pv >= 1e-6f);
#pragma unroll
    for (int dy = -1; dy <= 1; ++dy)
#pragma unroll
      for (int dx = -1; dx <= 1; ++dx) {
        if (dy == 0 && dx == 0) continue;
        int nh = hh + dy, nw = ww + dx;
        float pn = ((unsigned)nh < 96u && (unsigned)nw < 96u) ? pc[nh * Wn + nw] : 0.f;
        m = m && (pv >= pn);
      }
    best = fmaxf(best, pv + (m ? 1.f : 0.f));
  }
  score[(size_t)b * HW + pix] = best;
}

// iterative top-K selection, ties -> lower index; destructively marks score
__global__ __launch_bounds__(256) void qp_topk(float* __restrict__ score,
                                               int* __restrict__ idxout,
                                               int K, int off) {
  int b = blockIdx.x;
  float* s = score + (size_t)b * HW;
  __shared__ float sv[256];
  __shared__ int   si[256];
  for (int k = 0; k < K; ++k) {
    float best = -1e30f; int bi = 0x7fffffff;
    for (int p = threadIdx.x; p < HW; p += 256) {
      float v = s[p];
      if (v > best || (v == best && p < bi)) { best = v; bi = p; }
    }
    sv[threadIdx.x] = best; si[threadIdx.x] = bi;
    __syncthreads();
    for (int o = 128; o > 0; o >>= 1) {
      if (threadIdx.x < o) {
        float v2 = sv[threadIdx.x + o]; int i2 = si[threadIdx.x + o];
        if (v2 > sv[threadIdx.x] || (v2 == sv[threadIdx.x] && i2 < si[threadIdx.x])) {
          sv[threadIdx.x] = v2; si[threadIdx.x] = i2;
        }
      }
      __syncthreads();
    }
    if (threadIdx.x == 0) { idxout[b * NQn + off + k] = si[0]; s[si[0]] = -1e38f; }
    __syncthreads();
  }
}

// mean over W: [B,32,H,W] f32 -> [B,32,H]
__global__ void qp_mean_w(const float* __restrict__ h2, float* __restrict__ hm) {
  int e = blockIdx.x * blockDim.x + threadIdx.x;
  if (e >= Bn * 32 * Hn) return;
  const float* r = h2 + (size_t)e * Wn;
  float a = 0.f;
  for (int w = 0; w < Wn; ++w) a += r[w];
  hm[e] = a * (1.f / Wn);
}

// adaptive avg pool 96 -> 778 (torch semantics)
__global__ void qp_adapool(const float* __restrict__ hm, float* __restrict__ pool) {
  int e = blockIdx.x * blockDim.x + threadIdx.x;
  if (e >= Bn * 32 * 778) return;
  int i  = e % 778;
  int bc = e / 778;               // (b*32 + ch)
  int s  = (i * Hn) / 778;
  int t  = ((i + 1) * Hn + 777) / 778;
  float a = 0.f;
  for (int k = s; k < t; ++k) a += hm[(size_t)bc * Hn + k];
  pool[e] = a / (float)(t - s);
}

// cm[b][n] = pool[b] . cm_lw[n] + cm_lb[n]; also writes contact_map output
__global__ __launch_bounds__(256) void qp_cm_gemm(const float* __restrict__ pool,
                                                  const float* __restrict__ W,
                                                  const float* __restrict__ bias,
                                                  float* __restrict__ cm,
                                                  float* __restrict__ cmap_out) {
  const int n = blockIdx.x;  // 0..1555
  __shared__ float red[256];
  float accv[Bn];
#pragma unroll
  for (int b = 0; b < Bn; ++b) accv[b] = 0.f;
  const float* wr = W + (size_t)n * 24896;
  for (int k = threadIdx.x; k < 24896; k += 256) {
    float wv = wr[k];
#pragma unroll
    for (int b = 0; b < Bn; ++b) accv[b] += pool[(size_t)b * 24896 + k] * wv;
  }
  for (int b = 0; b < Bn; ++b) {
    red[threadIdx.x] = accv[b];
    __syncthreads();
    for (int o = 128; o > 0; o >>= 1) {
      if (threadIdx.x < o) red[threadIdx.x] += red[threadIdx.x + o];
      __syncthreads();
    }
    if (threadIdx.x == 0) {
      float r = red[0] + bias[n];
      cm[(size_t)b * 1556 + n] = r;
      cmap_out[(size_t)b * 1556 + n] = r;
    }
    __syncthreads();
  }
}

// mean over t of conv1d == conv of window sums: S[ic][j] = sum_t cm[ic][t+j]
__global__ __launch_bounds__(256) void qp_conv1d_mean(const float* __restrict__ cm,
                                                      const float* __restrict__ w,
                                                      const float* __restrict__ bc,
                                                      float* __restrict__ cc) {
  const int b = blockIdx.x;
  __shared__ float S[6];
  __shared__ float red[256];
  for (int combo = 0; combo < 6; ++combo) {
    int ic = combo / 3, j = combo % 3;
    float a = 0.f;
    for (int t = threadIdx.x; t < 776; t += 256)
      a += cm[(size_t)b * 1556 + ic * 778 + t + j];
    red[threadIdx.x] = a;
    __syncthreads();
    for (int o = 128; o > 0; o >>= 1) {
      if (threadIdx.x < o) red[threadIdx.x] += red[threadIdx.x + o];
      __syncthreads();
    }
    if (threadIdx.x == 0) S[combo] = red[0];
    __syncthreads();
  }
  const int co = threadIdx.x;  // 0..255
  float a = 0.f;
#pragma unroll
  for (int combo = 0; combo < 6; ++combo) {
    int ic = combo / 3, j = combo % 3;
    a += w[(co * 2 + ic) * 3 + j] * S[combo];
  }
  cc[(size_t)b * Cn + co] = a * (1.f / 776.f) + bc[co];
}

// feat[b][n] = cc[b] . q_lw[n] + q_lb[n], n < 25600 (grid 100 x 256)
__global__ __launch_bounds__(256) void qp_feat_gemm(const float* __restrict__ cc,
                                                    const float* __restrict__ qw,
                                                    const float* __restrict__ qb,
                                                    float* __restrict__ feat) {
  __shared__ float ccs[Bn * Cn];
  for (int k = threadIdx.x; k < Bn * Cn; k += 256) ccs[k] = cc[k];
  __syncthreads();
  const int n = blockIdx.x * 256 + threadIdx.x;
  float a[Bn];
#pragma unroll
  for (int b = 0; b < Bn; ++b) a[b] = 0.f;
  const float* w = qw + (size_t)n * Cn;
  for (int k = 0; k < Cn; ++k) {
    float wv = w[k];
#pragma unroll
    for (int b = 0; b < Bn; ++b) a[b] += ccs[b * Cn + k] * wv;
  }
  const float bb = qb[n];
#pragma unroll
  for (int b = 0; b < Bn; ++b) feat[(size_t)b * 25600 + n] = a[b] + bb;
}

// fused gather: proposals = x[.,.,idx] + feat ; pos = pos_emb[.,idx]
__global__ void qp_gather(const float* __restrict__ x, const float* __restrict__ pos,
                          const int* __restrict__ idx, const float* __restrict__ feat,
                          float* __restrict__ prop_out, float* __restrict__ pos_out) {
  int e = blockIdx.x * blockDim.x + threadIdx.x;
  if (e >= Bn * Cn * NQn) return;
  int q = e % NQn;
  int c = (e / NQn) % Cn;
  int b = e / (Cn * NQn);
  int id = idx[b * NQn + q];
  prop_out[e] = x[((size_t)b * Cn + c) * HW + id] + feat[(size_t)b * 25600 + c * NQn + q];
  pos_out[e]  = pos[(size_t)c * HW + id];
}

// ---------------------------------------------------------------------------
extern "C" void kernel_launch(void* const* d_in, const int* in_sizes, int n_in,
                              void* d_out, int out_size, void* d_ws, size_t ws_size,
                              hipStream_t stream) {
  (void)in_sizes; (void)n_in; (void)out_size; (void)ws_size;
  const float* x    = (const float*)d_in[0];
  const float* y    = (const float*)d_in[1];
  const float* pos  = (const float*)d_in[2];
  const float* lw1  = (const float*)d_in[3];  const float* lb1 = (const float*)d_in[4];
  const float* lw2  = (const float*)d_in[5];  const float* lb2 = (const float*)d_in[6];
  const float* rw1  = (const float*)d_in[7];  const float* rb1 = (const float*)d_in[8];
  const float* rw2  = (const float*)d_in[9];  const float* rb2 = (const float*)d_in[10];
  const float* ow1  = (const float*)d_in[11]; const float* ob1 = (const float*)d_in[12];
  const float* ow2  = (const float*)d_in[13]; const float* ob2 = (const float*)d_in[14];
  const float* cw1  = (const float*)d_in[15]; const float* cb1 = (const float*)d_in[16];
  const float* cw2  = (const float*)d_in[17]; const float* cb2 = (const float*)d_in[18];
  const float* cmlw = (const float*)d_in[19]; const float* cmlb = (const float*)d_in[20];
  const float* c1dw = (const float*)d_in[21]; const float* c1db = (const float*)d_in[22];
  const float* qlw  = (const float*)d_in[23]; const float* qlb  = (const float*)d_in[24];

  float* out    = (float*)d_out;
  float* o_prop = out;                 // [8,256,100]
  float* o_pos  = out + 204800;        // [8,256,100]
  float* o_left = out + 409600;        // [8,2,96,96]
  float* o_right= out + 557056;        // [8,2,96,96]
  float* o_obj  = out + 704512;        // [8,11,96,96]
  float* o_cmap = out + 1515520;       // [8,2,778,1]

  char* wsp = (char*)d_ws;
  auto alloc = [&](size_t bytes) -> char* {
    char* p = wsp;
    wsp += (bytes + 255) & ~(size_t)255;
    return p;
  };
  const size_t NXB = (size_t)Bn * Hn * Wn * Cn;  // 18,874,368
  unsigned short* xb   = (unsigned short*)alloc(NXB * 2);
  unsigned short* yb   = (unsigned short*)alloc(NXB * 2);
  unsigned short* wfL  = (unsigned short*)alloc((size_t)9 * 256 * 256 * 2);
  unsigned short* wfR  = (unsigned short*)alloc((size_t)9 * 256 * 256 * 2);
  unsigned short* wfO  = (unsigned short*)alloc((size_t)9 * 256 * 256 * 2);
  unsigned short* wfC1 = (unsigned short*)alloc((size_t)9 * 256 * 64 * 2);
  unsigned short* wfC2 = (unsigned short*)alloc((size_t)9 * 64 * 32 * 2);
  unsigned short* h1   = (unsigned short*)alloc((size_t)Bn * Hn * Wn * 64 * 2);
  float* h2    = (float*)alloc((size_t)Bn * 32 * Hn * Wn * 4);
  float* pL    = (float*)alloc((size_t)Bn * 2 * HW * 4);
  float* pR    = (float*)alloc((size_t)Bn * 2 * HW * 4);
  float* pO    = (float*)alloc((size_t)Bn * OBJ * HW * 4);
  float* sL    = (float*)alloc((size_t)Bn * HW * 4);
  float* sR    = (float*)alloc((size_t)Bn * HW * 4);
  float* sO    = (float*)alloc((size_t)Bn * HW * 4);
  int*   idx   = (int*)alloc((size_t)Bn * NQn * 4);
  float* hm    = (float*)alloc((size_t)Bn * 32 * Hn * 4);
  float* poolb = (float*)alloc((size_t)Bn * 32 * 778 * 4);
  float* cmbuf = (float*)alloc((size_t)Bn * 1556 * 4);
  float* ccbuf = (float*)alloc((size_t)Bn * Cn * 4);
  float* featb = (float*)alloc((size_t)Bn * 25600 * 4);

  const int T = 256;
  // --- pack activations & weights ---
  const int totX = (int)NXB;
  qp_nchw_to_nhwc_bf16<<<(totX + T - 1) / T, T, 0, stream>>>(x, xb, 256, totX);
  qp_nchw_to_nhwc_bf16<<<(totX + T - 1) / T, T, 0, stream>>>(y, yb, 256, totX);
  qp_pack_wfrag<<<(9 * 256 * 256 + T - 1) / T, T, 0, stream>>>(lw1, wfL, 256, 256, 9 * 256 * 256);
  qp_pack_wfrag<<<(9 * 256 * 256 + T - 1) / T, T, 0, stream>>>(rw1, wfR, 256, 256, 9 * 256 * 256);
  qp_pack_wfrag<<<(9 * 256 * 256 + T - 1) / T, T, 0, stream>>>(ow1, wfO, 256, 256, 9 * 256 * 256);
  qp_pack_wfrag<<<(9 * 256 * 64  + T - 1) / T, T, 0, stream>>>(cw1, wfC1, 256, 64, 9 * 256 * 64);
  qp_pack_wfrag<<<(9 * 64 * 32   + T - 1) / T, T, 0, stream>>>(cw2, wfC2, 64, 32, 9 * 64 * 32);

  // --- WMMA convs ---
  dim3 cgrid(Hn * (Wn / 32), Bn);  // (288, 8)
  qp_conv3x3_wmma<256, 256, 0, 2 ><<<cgrid, 256, 0, stream>>>(xb, wfL, lb1, lw2, lb2, o_left);
  qp_conv3x3_wmma<256, 256, 0, 2 ><<<cgrid, 256, 0, stream>>>(xb, wfR, rb1, rw2, rb2, o_right);
  qp_conv3x3_wmma<256, 256, 0, 11><<<cgrid, 256, 0, stream>>>(xb, wfO, ob1, ow2, ob2, o_obj);
  qp_conv3x3_wmma<256, 64,  1, 1 ><<<cgrid, 256, 0, stream>>>(yb, wfC1, cb1, nullptr, nullptr, h1);
  qp_conv3x3_wmma<64,  32,  2, 1 ><<<cgrid, 256, 0, stream>>>(h1, wfC2, cb2, nullptr, nullptr, h2);

  // --- softmax / mask / NMS / top-k ---
  const int tp = Bn * HW;
  qp_softmax_mask<2 ><<<(tp + T - 1) / T, T, 0, stream>>>(o_left, pL);
  qp_softmax_mask<2 ><<<(tp + T - 1) / T, T, 0, stream>>>(o_right, pR);
  qp_softmax_mask<11><<<(tp + T - 1) / T, T, 0, stream>>>(o_obj, pO);
  qp_nms_score<2, 1  ><<<(tp + T - 1) / T, T, 0, stream>>>(pL, sL);
  qp_nms_score<2, 1  ><<<(tp + T - 1) / T, T, 0, stream>>>(pR, sR);
  qp_nms_score<11, 10><<<(tp + T - 1) / T, T, 0, stream>>>(pO, sO);
  qp_topk<<<Bn, 256, 0, stream>>>(sL, idx, 40, 0);
  qp_topk<<<Bn, 256, 0, stream>>>(sR, idx, 40, 40);
  qp_topk<<<Bn, 256, 0, stream>>>(sO, idx, 20, 80);

  // --- contact branch ---
  qp_mean_w<<<(Bn * 32 * Hn + T - 1) / T, T, 0, stream>>>(h2, hm);
  qp_adapool<<<(Bn * 32 * 778 + T - 1) / T, T, 0, stream>>>(hm, poolb);
  qp_cm_gemm<<<1556, 256, 0, stream>>>(poolb, cmlw, cmlb, cmbuf, o_cmap);
  qp_conv1d_mean<<<Bn, 256, 0, stream>>>(cmbuf, c1dw, c1db, ccbuf);
  qp_feat_gemm<<<100, 256, 0, stream>>>(ccbuf, qlw, qlb, featb);

  // --- final gather + add ---
  qp_gather<<<(Bn * Cn * NQn + T - 1) / T, T, 0, stream>>>(x, pos, idx, featb, o_prop, o_pos);
}
